// TransformerLM_72971494359535
// MI455X (gfx1250) — matrix operand order repaired and checked by reading, not compiled, not run on hardware
//
#include <hip/hip_runtime.h>

// ---------------- problem constants ----------------
#define Bb  2
#define Tt  2048
#define Cc  1024
#define Hh  16
#define KVh 4
#define HD  64
#define Ee  8
#define FEe 1024
#define FSs 2048
#define NT  (Bb*Tt)          // 4096 tokens

typedef __bf16 bf16;
typedef __attribute__((ext_vector_type(16))) __bf16 v16bf;
typedef __attribute__((ext_vector_type(8)))  float  v8f;

union FragBF { unsigned int u[8]; v16bf v; };

static __device__ __forceinline__ bf16 f2bf(float f) { return (bf16)f; }

// =====================================================================
// fp32 [batch][K][N] -> bf16 [batch][N][K]  (transposing convert, so GEMM
// B-tiles are K-contiguous and load with b128 vector ops)
// =====================================================================
__global__ void cvt_transpose_kernel(const float* __restrict__ s, bf16* __restrict__ d,
                                     int K, int N, int total)
{
    int gid = blockIdx.x * blockDim.x + threadIdx.x;
    if (gid >= total) return;
    int n = gid % N;
    int k = (gid / N) % K;
    int b = gid / (N * K);
    d[((size_t)b * N + n) * K + k] = f2bf(s[gid]);
}

__global__ void zero_u32_kernel(unsigned int* __restrict__ p, int n) {
    int i = blockIdx.x * blockDim.x + threadIdx.x;
    if (i < n) p[i] = 0u;
}

// =====================================================================
// RMSNorm (optionally fused residual add). Writes x1 fp32, h fp32, h bf16.
// =====================================================================
__global__ __launch_bounds__(256) void rmsnorm_kernel(
    const float* __restrict__ x, const float* __restrict__ res,
    const float* __restrict__ w, float* __restrict__ x1out,
    float* __restrict__ hfout, bf16* __restrict__ hbout)
{
    const int row = blockIdx.x;
    const int tid = threadIdx.x;
    const size_t base = (size_t)row * Cc;
    float vals[4];
    float ss = 0.f;
#pragma unroll
    for (int i = 0; i < 4; ++i) {
        int c = tid + i * 256;
        float v = x[base + c];
        if (res) v += res[base + c];
        vals[i] = v; ss += v * v;
    }
#pragma unroll
    for (int off = 16; off > 0; off >>= 1) ss += __shfl_xor(ss, off);
    __shared__ float red[8];
    if ((tid & 31) == 0) red[tid >> 5] = ss;
    __syncthreads();
    float tot = 0.f;
#pragma unroll
    for (int i = 0; i < 8; ++i) tot += red[i];
    const float rs = rsqrtf(tot * (1.0f / (float)Cc) + 1e-6f);
#pragma unroll
    for (int i = 0; i < 4; ++i) {
        int c = tid + i * 256;
        float hv = vals[i] * rs * w[c];
        if (x1out) x1out[base + c] = vals[i];
        if (hfout) hfout[base + c] = hv;
        hbout[base + c] = f2bf(hv);
    }
}

// =====================================================================
// Shared WMMA main loop.  A: [M][K] row-major.  Bt: [N][K] row-major
// (pre-transposed weights).  Tile 128x128, 256 threads / 8 waves,
// each wave 32x64 = 2x4 wmma tiles, K-steps of 32.
// aRow/bRow: this thread's global source row pointers (at k=0).
// =====================================================================
__device__ __forceinline__ void wmma_mainloop(
    const bf16* __restrict__ aRow, const bf16* __restrict__ bRow, int K,
    bf16 (* __restrict__ As)[32], bf16 (* __restrict__ Bs)[32],
    int tid, int wm, int wn, int hl, int l15, v8f acc[2][4])
{
    const int row  = tid >> 1;
    const int half = tid & 1;
    for (int k0 = 0; k0 < K; k0 += 32) {
        {
            const uint4* sa = (const uint4*)(aRow + k0 + half * 16);
            uint4* da = (uint4*)(&As[row][half * 16]);
            da[0] = sa[0]; da[1] = sa[1];
            const uint4* sb = (const uint4*)(bRow + k0 + half * 16);
            uint4* db = (uint4*)(&Bs[row][half * 16]);
            db[0] = sb[0]; db[1] = sb[1];
        }
        if (k0 + 32 < K) {
            __builtin_prefetch((const void*)(aRow + k0 + 32 + half * 16), 0, 3);
            __builtin_prefetch((const void*)(bRow + k0 + 32 + half * 16), 0, 3);
        }
        __syncthreads();

        FragBF a[2], b[4];
#pragma unroll
        for (int r = 0; r < 2; ++r) {
            const unsigned int* ap = (const unsigned int*)(&As[wm * 32 + r * 16 + l15][0]);
#pragma unroll
            for (int d = 0; d < 4; ++d) {
                a[r].u[d]     = ap[hl * 4 + d];
                a[r].u[d + 4] = ap[8 + hl * 4 + d];
            }
        }
#pragma unroll
        for (int c = 0; c < 4; ++c) {
            const unsigned int* bp = (const unsigned int*)(&Bs[wn * 64 + c * 16 + l15][0]);
#pragma unroll
            for (int d = 0; d < 4; ++d) { b[c].u[d] = bp[hl * 4 + d]; b[c].u[d + 4] = bp[8 + hl * 4 + d]; }
        }
#pragma unroll
        for (int r = 0; r < 2; ++r)
#pragma unroll
            for (int c = 0; c < 4; ++c)
                acc[r][c] = __builtin_amdgcn_wmma_f32_16x16x32_bf16(
                    false, a[r].v, false, b[c].v, (short)0, acc[r][c], false, false);
        __syncthreads();
    }
}

// C[M,N] (+)= A[M,K] * Bt[N,K]^T ; BETA is compile-time
template <int BETA>
__global__ __launch_bounds__(256) void gemm_bf16_kernel(
    const bf16* __restrict__ A, const bf16* __restrict__ Bt,
    float* __restrict__ C, int N, int K)
{
    __shared__ bf16 As[128][32];
    __shared__ bf16 Bs[128][32];
    const int tid  = threadIdx.x;
    const int lane = tid & 31, wave = tid >> 5;
    const int hl = lane >> 4, l15 = lane & 15;
    const int wm = wave >> 1, wn = wave & 1;
    const int m0 = blockIdx.y * 128;
    const int n0 = blockIdx.x * 128;

    v8f acc[2][4];
#pragma unroll
    for (int r = 0; r < 2; ++r)
#pragma unroll
        for (int c = 0; c < 4; ++c)
#pragma unroll
            for (int i = 0; i < 8; ++i) acc[r][c][i] = 0.f;

    const bf16* aRow = A  + (size_t)(m0 + (tid >> 1)) * K;
    const bf16* bRow = Bt + (size_t)(n0 + (tid >> 1)) * K;
    wmma_mainloop(aRow, bRow, K, As, Bs, tid, wm, wn, hl, l15, acc);

#pragma unroll
    for (int r = 0; r < 2; ++r)
#pragma unroll
        for (int c = 0; c < 4; ++c)
#pragma unroll
            for (int v = 0; v < 8; ++v) {
                int gr = m0 + wm * 32 + r * 16 + hl * 8 + v;
                int gc = n0 + wn * 64 + c * 16 + l15;
                if (BETA) C[(size_t)gr * N + gc] += acc[r][c][v];
                else      C[(size_t)gr * N + gc]  = acc[r][c][v];
            }
}

// Gathered GEMM for MoE: C[slot,N] = A[idx[slot],K] * Bt^T   (slots < *cnt)
__global__ __launch_bounds__(256) void gemm_bf16_gather_kernel(
    const bf16* __restrict__ A, const bf16* __restrict__ Bt,
    float* __restrict__ C, const int* __restrict__ idx,
    const int* __restrict__ cnt, int N, int K)
{
    const int m0 = blockIdx.y * 128;
    const int count = *cnt;
    if (m0 >= count) return;
    __shared__ bf16 As[128][32];
    __shared__ bf16 Bs[128][32];
    const int tid  = threadIdx.x;
    const int lane = tid & 31, wave = tid >> 5;
    const int hl = lane >> 4, l15 = lane & 15;
    const int wm = wave >> 1, wn = wave & 1;
    const int n0 = blockIdx.x * 128;

    v8f acc[2][4];
#pragma unroll
    for (int r = 0; r < 2; ++r)
#pragma unroll
        for (int c = 0; c < 4; ++c)
#pragma unroll
            for (int i = 0; i < 8; ++i) acc[r][c][i] = 0.f;

    int slot = m0 + (tid >> 1);
    int arow_g = (slot < count) ? idx[slot] : idx[0];
    const bf16* aRow = A  + (size_t)arow_g * K;
    const bf16* bRow = Bt + (size_t)(n0 + (tid >> 1)) * K;
    wmma_mainloop(aRow, bRow, K, As, Bs, tid, wm, wn, hl, l15, acc);

#pragma unroll
    for (int r = 0; r < 2; ++r)
#pragma unroll
        for (int c = 0; c < 4; ++c)
#pragma unroll
            for (int v = 0; v < 8; ++v) {
                int gr = m0 + wm * 32 + r * 16 + hl * 8 + v;
                int gc = n0 + wn * 64 + c * 16 + l15;
                C[(size_t)gr * N + gc] = acc[r][c][v];
            }
}

// Scatter-add GEMM for MoE down-proj: Y[idx[m],:] += wgt[m] * (A[m,:]*Bt^T)
__global__ __launch_bounds__(256) void gemm_bf16_scatter_kernel(
    const bf16* __restrict__ A, const bf16* __restrict__ Bt,
    float* __restrict__ Y, const int* __restrict__ idx,
    const float* __restrict__ wgt, const int* __restrict__ cnt, int N, int K)
{
    const int m0 = blockIdx.y * 128;
    const int count = *cnt;
    if (m0 >= count) return;
    __shared__ bf16 As[128][32];
    __shared__ bf16 Bs[128][32];
    const int tid  = threadIdx.x;
    const int lane = tid & 31, wave = tid >> 5;
    const int hl = lane >> 4, l15 = lane & 15;
    const int wm = wave >> 1, wn = wave & 1;
    const int n0 = blockIdx.x * 128;

    v8f acc[2][4];
#pragma unroll
    for (int r = 0; r < 2; ++r)
#pragma unroll
        for (int c = 0; c < 4; ++c)
#pragma unroll
            for (int i = 0; i < 8; ++i) acc[r][c][i] = 0.f;

    const bf16* aRow = A  + (size_t)(m0 + (tid >> 1)) * K;
    const bf16* bRow = Bt + (size_t)(n0 + (tid >> 1)) * K;
    wmma_mainloop(aRow, bRow, K, As, Bs, tid, wm, wn, hl, l15, acc);

#pragma unroll
    for (int r = 0; r < 2; ++r)
#pragma unroll
        for (int c = 0; c < 4; ++c)
#pragma unroll
            for (int v = 0; v < 8; ++v) {
                int m = m0 + wm * 32 + r * 16 + hl * 8 + v;
                if (m < count) {
                    int tok = idx[m];
                    int gc  = n0 + wn * 64 + c * 16 + l15;
                    Y[(size_t)tok * N + gc] += wgt[m] * acc[r][c][v];
                }
            }
}

// =====================================================================
// RoPE + relayout  [B*T, nh*64] fp32 -> bf16
//   t_major==0: [B, nh, T, 64]   (Q, K)
//   t_major==1: [B, nh, 64, T]   (V; matches attention B-fragment layout)
// =====================================================================
__global__ void rope_cvt_kernel(const float* __restrict__ src, bf16* __restrict__ dst,
                                int nh, int do_rope, int t_major)
{
    int gid = blockIdx.x * blockDim.x + threadIdx.x;
    int total = NT * nh * 32;
    if (gid >= total) return;
    int j = gid & 31;
    int h = (gid >> 5) % nh;
    int n = gid / (32 * nh);
    int t = n & (Tt - 1), b = n >> 11;
    const float* s = src + (size_t)n * (nh * HD) + h * HD + 2 * j;
    float x0 = s[0], x1 = s[1];
    if (do_rope) {
        float inv = __powf(10000.f, -(float)(2 * j) / (float)HD);
        float ang = (float)t * inv;
        float sn, cs;
        __sincosf(ang, &sn, &cs);
        float r0 = x0 * cs - x1 * sn;
        float r1 = x1 * cs + x0 * sn;
        x0 = r0; x1 = r1;
    }
    if (t_major) {
        bf16* d = dst + (((size_t)b * nh + h) * HD + 2 * j) * Tt + t;
        d[0]  = f2bf(x0);
        d[Tt] = f2bf(x1);
    } else {
        bf16* d = dst + (((size_t)b * nh + h) * Tt + t) * HD + 2 * j;
        d[0] = f2bf(x0); d[1] = f2bf(x1);
    }
}

// =====================================================================
// Flash attention (causal, GQA rep=4). 128 threads = 4 waves,
// each wave owns 16 query rows; 32-key tiles; S and P*V via WMMA.
// Q,K: [B,{H|KV},T,64] bf16.  V: [B,KV,64,T] bf16.  O: [B,T,H*64] bf16.
// =====================================================================
__global__ __launch_bounds__(128) void flash_attn_kernel(
    const bf16* __restrict__ Q, const bf16* __restrict__ Kd,
    const bf16* __restrict__ Vd, bf16* __restrict__ O)
{
    __shared__ bf16 Ks[32][64];      // [key][hd] -> B-frag layout for S
    __shared__ bf16 Vs[64][32];      // [hd][key] -> B-frag layout for P*V
    __shared__ bf16 Ps[4][16][32];   // per-wave P tile (D-layout -> A-layout)
    const int tid  = threadIdx.x;
    const int lane = tid & 31, wave = tid >> 5;
    const int hl = lane >> 4, l15 = lane & 15;
    const int bx = blockIdx.x;
    const int h  = blockIdx.y, b = blockIdx.z;
    const int kvh = h >> 2;
    const int q0 = bx * 64 + wave * 16;

    FragBF qa[2];
    {
        const unsigned int* qp =
            (const unsigned int*)(Q + (((size_t)b * Hh + h) * Tt + q0 + l15) * HD);
#pragma unroll
        for (int s = 0; s < 2; ++s)
#pragma unroll
            for (int d = 0; d < 4; ++d) {
                qa[s].u[d]     = qp[s * 16 + hl * 4 + d];
                qa[s].u[d + 4] = qp[s * 16 + 8 + hl * 4 + d];
            }
    }

    float mrow[8], lrow[8];
    v8f oacc[4];
#pragma unroll
    for (int v = 0; v < 8; ++v) { mrow[v] = -1e30f; lrow[v] = 0.f; }
#pragma unroll
    for (int t = 0; t < 4; ++t)
#pragma unroll
        for (int i = 0; i < 8; ++i) oacc[t][i] = 0.f;

    const bf16* kbase = Kd + ((size_t)b * KVh + kvh) * Tt * HD;   // [key][hd]
    const bf16* vbase = Vd + ((size_t)b * KVh + kvh) * HD * Tt;   // [hd][key]
    const int nkb = (bx + 1) * 2;

    for (int kb = 0; kb < nkb; ++kb) {
        const int kt0 = kb * 32;
        { // K tile: natural copy; V tile: rows are hd, keys contiguous
            int key = tid >> 2, seg = (tid & 3) * 16;
            const uint4* ks = (const uint4*)(kbase + (size_t)(kt0 + key) * HD + seg);
            uint4* kd = (uint4*)(&Ks[key][seg]);
            kd[0] = ks[0]; kd[1] = ks[1];
            int hd = tid >> 1, vh = (tid & 1) * 16;
            const uint4* vs = (const uint4*)(vbase + (size_t)hd * Tt + kt0 + vh);
            uint4* vd = (uint4*)(&Vs[hd][vh]);
            vd[0] = vs[0]; vd[1] = vs[1];
            if (kb + 1 < nkb) {
                __builtin_prefetch((const void*)(kbase + (size_t)(kt0 + 32 + key) * HD + seg), 0, 3);
                __builtin_prefetch((const void*)(vbase + (size_t)hd * Tt + kt0 + 32 + vh), 0, 3);
            }
        }
        __syncthreads();

        // S = Q * K^T : two 16x16 tiles (32 keys), accumulated over HD=64
        v8f sacc[2];
#pragma unroll
        for (int c = 0; c < 2; ++c)
#pragma unroll
            for (int i = 0; i < 8; ++i) sacc[c][i] = 0.f;
#pragma unroll
        for (int c = 0; c < 2; ++c) {
            const unsigned int* kp = (const unsigned int*)(&Ks[c * 16 + l15][0]);
#pragma unroll
            for (int s = 0; s < 2; ++s) {
                FragBF kf;
#pragma unroll
                for (int d = 0; d < 4; ++d) { kf.u[d] = kp[s * 16 + hl * 4 + d]; kf.u[d + 4] = kp[s * 16 + 8 + hl * 4 + d]; }
                sacc[c] = __builtin_amdgcn_wmma_f32_16x16x32_bf16(
                    false, qa[s].v, false, kf.v, (short)0, sacc[c], false, false);
            }
        }

        // online softmax
#pragma unroll
        for (int v = 0; v < 8; ++v) {
            int qr = q0 + hl * 8 + v;
            float s0 = sacc[0][v] * 0.125f;
            float s1 = sacc[1][v] * 0.125f;
            if (kt0 + l15 > qr)      s0 = -1e9f;
            if (kt0 + 16 + l15 > qr) s1 = -1e9f;
            float mx = fmaxf(s0, s1);
#pragma unroll
            for (int off = 8; off > 0; off >>= 1) mx = fmaxf(mx, __shfl_xor(mx, off));
            float mnew  = fmaxf(mrow[v], mx);
            float alpha = __expf(mrow[v] - mnew);
            float p0 = __expf(s0 - mnew);
            float p1 = __expf(s1 - mnew);
            float rs = p0 + p1;
#pragma unroll
            for (int off = 8; off > 0; off >>= 1) rs += __shfl_xor(rs, off);
            lrow[v] = lrow[v] * alpha + rs;
            mrow[v] = mnew;
            Ps[wave][hl * 8 + v][l15]      = f2bf(p0);
            Ps[wave][hl * 8 + v][16 + l15] = f2bf(p1);
#pragma unroll
            for (int t = 0; t < 4; ++t) oacc[t][v] *= alpha;
        }
        __syncthreads();

        // O += P * V
        FragBF pa;
        {
            const unsigned int* pp = (const unsigned int*)(&Ps[wave][l15][0]);
#pragma unroll
            for (int d = 0; d < 4; ++d) {
                pa.u[d]     = pp[hl * 4 + d];
                pa.u[d + 4] = pp[8 + hl * 4 + d];
            }
        }
#pragma unroll
        for (int t = 0; t < 4; ++t) {
            FragBF vf;
            const unsigned int* vp = (const unsigned int*)(&Vs[t * 16 + l15][0]);
#pragma unroll
            for (int d = 0; d < 4; ++d) { vf.u[d] = vp[hl * 4 + d]; vf.u[d + 4] = vp[8 + hl * 4 + d]; }
            oacc[t] = __builtin_amdgcn_wmma_f32_16x16x32_bf16(
                false, pa.v, false, vf.v, (short)0, oacc[t], false, false);
        }
        __syncthreads();
    }

#pragma unroll
    for (int v = 0; v < 8; ++v) {
        float inv = 1.f / lrow[v];
        int qr = q0 + hl * 8 + v;
        bf16* orow = O + ((size_t)b * Tt + qr) * Cc + h * HD;
#pragma unroll
        for (int t = 0; t < 4; ++t) orow[t * 16 + l15] = f2bf(oacc[t][v] * inv);
    }
}

// =====================================================================
// Gate logits: glog[n,e] = h2[n,:] . gate_w[:,e]
// =====================================================================
__global__ void gate_kernel(const float* __restrict__ h2, const float* __restrict__ gw,
                            float* __restrict__ glog)
{
    int gid = blockIdx.x * blockDim.x + threadIdx.x;
    if (gid >= NT * Ee) return;
    int n = gid >> 3, e = gid & 7;
    const float* hr = h2 + (size_t)n * Cc;
    float s = 0.f;
    for (int i = 0; i < Cc; ++i) s += hr[i] * gw[i * Ee + e];
    glog[gid] = s;
}

// softmax + top-2 routing; builds per-expert token lists
__global__ void topk_kernel(const float* __restrict__ glog, int* __restrict__ cnt,
                            int* __restrict__ idx, float* __restrict__ wgt)
{
    int n = blockIdx.x * blockDim.x + threadIdx.x;
    if (n >= NT) return;
    float l[Ee], p[Ee];
    float mx = -1e30f;
#pragma unroll
    for (int e = 0; e < Ee; ++e) { l[e] = glog[n * Ee + e]; mx = fmaxf(mx, l[e]); }
    float se = 0.f;
#pragma unroll
    for (int e = 0; e < Ee; ++e) { p[e] = __expf(l[e] - mx); se += p[e]; }
    int i1 = 0;
#pragma unroll
    for (int e = 1; e < Ee; ++e) if (p[e] > p[i1]) i1 = e;
    int i2 = (i1 == 0) ? 1 : 0;
#pragma unroll
    for (int e = 0; e < Ee; ++e) if (e != i1 && p[e] > p[i2]) i2 = e;
    float w1 = p[i1] / se, w2 = p[i2] / se;
    int pos = atomicAdd(&cnt[i1], 1); idx[i1 * NT + pos] = n; wgt[i1 * NT + pos] = w1;
    pos = atomicAdd(&cnt[i2], 1);     idx[i2 * NT + pos] = n; wgt[i2 * NT + pos] = w2;
}

// he = silu(a) * b, cast bf16
__global__ void silu_mul_kernel(const float* __restrict__ a, const float* __restrict__ b,
                                bf16* __restrict__ o, int n)
{
    int i = blockIdx.x * blockDim.x + threadIdx.x;
    if (i < n) {
        float x = a[i];
        float s = x / (1.f + __expf(-x));
        o[i] = f2bf(s * b[i]);
    }
}

// out = x1 + y
__global__ void final_add_kernel(const float* __restrict__ x1, const float* __restrict__ y,
                                 float* __restrict__ out, int n)
{
    int i = blockIdx.x * blockDim.x + threadIdx.x;
    if (i < n) out[i] = x1[i] + y[i];
}

// =====================================================================
// Host-side orchestration
// =====================================================================
static inline char* bump(char* ws, size_t& off, size_t bytes) {
    char* p = ws + off;
    off += (bytes + 255) & ~(size_t)255;
    return p;
}

extern "C" void kernel_launch(void* const* d_in, const int* in_sizes, int n_in,
                              void* d_out, int out_size, void* d_ws, size_t ws_size,
                              hipStream_t stream) {
    (void)in_sizes; (void)n_in; (void)out_size; (void)ws_size;
    const float* x   = (const float*)d_in[0];
    const float* n1w = (const float*)d_in[1];
    const float* n2w = (const float*)d_in[2];
    const float* wq  = (const float*)d_in[3];
    const float* wk  = (const float*)d_in[4];
    const float* wv  = (const float*)d_in[5];
    const float* wo  = (const float*)d_in[6];
    const float* gw  = (const float*)d_in[7];
    const float* ew1 = (const float*)d_in[8];
    const float* ew2 = (const float*)d_in[9];
    const float* ewp = (const float*)d_in[10];
    const float* sw1 = (const float*)d_in[11];
    const float* sw2 = (const float*)d_in[12];
    const float* swp = (const float*)d_in[13];

    float* out  = (float*)d_out;
    float* glog = out + (size_t)NT * Cc;     // gate_logits tail of the tuple

    char* ws = (char*)d_ws;
    size_t off = 0;
    bf16* wq_b = (bf16*)bump(ws, off, (size_t)Cc * Cc * 2);            // [N=C][K=C]
    bf16* wk_b = (bf16*)bump(ws, off, (size_t)Cc * (KVh * HD) * 2);    // [256][1024]
    bf16* wv_b = (bf16*)bump(ws, off, (size_t)Cc * (KVh * HD) * 2);
    bf16* wo_b = (bf16*)bump(ws, off, (size_t)Cc * Cc * 2);
    bf16* w1_b = (bf16*)bump(ws, off, (size_t)Ee * Cc * FEe * 2);      // [e][FE][C]
    bf16* w2_b = (bf16*)bump(ws, off, (size_t)Ee * Cc * FEe * 2);
    bf16* wp_b = (bf16*)bump(ws, off, (size_t)Ee * FEe * Cc * 2);      // [e][C][FE]
    bf16* s1_b = (bf16*)bump(ws, off, (size_t)Cc * FSs * 2);           // [FS][C]
    bf16* s2_b = (bf16*)bump(ws, off, (size_t)Cc * FSs * 2);
    bf16* sp_b = (bf16*)bump(ws, off, (size_t)FSs * Cc * 2);           // [C][FS]
    bf16* h1_b = (bf16*)bump(ws, off, (size_t)NT * Cc * 2);
    float* qf  = (float*)bump(ws, off, (size_t)NT * Cc * 4);           // reused as attn-out
    float* kf  = (float*)bump(ws, off, (size_t)NT * KVh * HD * 4);
    float* vf  = (float*)bump(ws, off, (size_t)NT * KVh * HD * 4);
    bf16* q_b  = (bf16*)bump(ws, off, (size_t)NT * Cc * 2);
    bf16* k_b  = (bf16*)bump(ws, off, (size_t)NT * KVh * HD * 2);
    bf16* v_b  = (bf16*)bump(ws, off, (size_t)NT * KVh * HD * 2);
    bf16* o_b  = (bf16*)bump(ws, off, (size_t)NT * Cc * 2);
    float* x1  = (float*)bump(ws, off, (size_t)NT * Cc * 4);
    float* h2f = (float*)bump(ws, off, (size_t)NT * Cc * 4);
    bf16* h2b  = (bf16*)bump(ws, off, (size_t)NT * Cc * 2);
    int*   cnt = (int*)bump(ws, off, Ee * 4);
    int*   idx = (int*)bump(ws, off, (size_t)Ee * NT * 4);
    float* wgt = (float*)bump(ws, off, (size_t)Ee * NT * 4);
    float* t1  = (float*)bump(ws, off, (size_t)NT * FSs * 4);
    float* t2  = (float*)bump(ws, off, (size_t)NT * FSs * 4);
    bf16* heb  = (bf16*)bump(ws, off, (size_t)NT * FSs * 2);
    float* yac = (float*)bump(ws, off, (size_t)NT * Cc * 4);

    auto cvtT = [&](const float* s, bf16* d, int batch, int K, int N) {
        int total = batch * K * N;
        cvt_transpose_kernel<<<(total + 255) / 256, 256, 0, stream>>>(s, d, K, N, total);
    };
    // weights -> bf16, pre-transposed to [N][K]
    cvtT(wq, wq_b, 1, Cc, Cc);
    cvtT(wk, wk_b, 1, Cc, KVh * HD);
    cvtT(wv, wv_b, 1, Cc, KVh * HD);
    cvtT(wo, wo_b, 1, Cc, Cc);
    cvtT(ew1, w1_b, Ee, Cc, FEe);
    cvtT(ew2, w2_b, Ee, Cc, FEe);
    cvtT(ewp, wp_b, Ee, FEe, Cc);
    cvtT(sw1, s1_b, 1, Cc, FSs);
    cvtT(sw2, s2_b, 1, Cc, FSs);
    cvtT(swp, sp_b, 1, FSs, Cc);

    // norm1
    rmsnorm_kernel<<<NT, 256, 0, stream>>>(x, nullptr, n1w, nullptr, nullptr, h1_b);

    // QKV projections
    gemm_bf16_kernel<0><<<dim3(Cc / 128, NT / 128), 256, 0, stream>>>(h1_b, wq_b, qf, Cc, Cc);
    gemm_bf16_kernel<0><<<dim3((KVh * HD) / 128, NT / 128), 256, 0, stream>>>(h1_b, wk_b, kf, KVh * HD, Cc);
    gemm_bf16_kernel<0><<<dim3((KVh * HD) / 128, NT / 128), 256, 0, stream>>>(h1_b, wv_b, vf, KVh * HD, Cc);

    // RoPE + relayout (V stored [B,KV,HD,T] for direct B-fragment tiles)
    rope_cvt_kernel<<<(NT * Hh * 32 + 255) / 256, 256, 0, stream>>>(qf, q_b, Hh, 1, 0);
    rope_cvt_kernel<<<(NT * KVh * 32 + 255) / 256, 256, 0, stream>>>(kf, k_b, KVh, 1, 0);
    rope_cvt_kernel<<<(NT * KVh * 32 + 255) / 256, 256, 0, stream>>>(vf, v_b, KVh, 0, 1);

    // attention
    flash_attn_kernel<<<dim3(Tt / 64, Hh, Bb), 128, 0, stream>>>(q_b, k_b, v_b, o_b);

    // output projection (into qf, reused)
    gemm_bf16_kernel<0><<<dim3(Cc / 128, NT / 128), 256, 0, stream>>>(o_b, wo_b, qf, Cc, Cc);

    // residual + norm2
    rmsnorm_kernel<<<NT, 256, 0, stream>>>(x, qf, n2w, x1, h2f, h2b);

    // gate logits straight into d_out tail, then routing
    gate_kernel<<<(NT * Ee + 255) / 256, 256, 0, stream>>>(h2f, gw, glog);
    zero_u32_kernel<<<1, 32, 0, stream>>>((unsigned int*)cnt, Ee);
    zero_u32_kernel<<<(NT * Cc + 255) / 256, 256, 0, stream>>>((unsigned int*)yac, NT * Cc);
    topk_kernel<<<(NT + 255) / 256, 256, 0, stream>>>(glog, cnt, idx, wgt);

    // MoE: top-2 routed experts only
    for (int e = 0; e < Ee; ++e) {
        const bf16* W1 = w1_b + (size_t)e * Cc * FEe;
        const bf16* W2 = w2_b + (size_t)e * Cc * FEe;
        const bf16* Wp = wp_b + (size_t)e * FEe * Cc;
        const int* ide = idx + (size_t)e * NT;
        const float* wge = wgt + (size_t)e * NT;
        gemm_bf16_gather_kernel<<<dim3(FEe / 128, NT / 128), 256, 0, stream>>>(h2b, W1, t1, ide, cnt + e, FEe, Cc);
        gemm_bf16_gather_kernel<<<dim3(FEe / 128, NT / 128), 256, 0, stream>>>(h2b, W2, t2, ide, cnt + e, FEe, Cc);
        silu_mul_kernel<<<(NT * FEe + 255) / 256, 256, 0, stream>>>(t1, t2, heb, NT * FEe);
        gemm_bf16_scatter_kernel<<<dim3(Cc / 128, NT / 128), 256, 0, stream>>>(heb, Wp, yac, ide, wge, cnt + e, Cc, FEe);
    }

    // shared FFN, accumulated into yac
    gemm_bf16_kernel<0><<<dim3(FSs / 128, NT / 128), 256, 0, stream>>>(h2b, s1_b, t1, FSs, Cc);
    gemm_bf16_kernel<0><<<dim3(FSs / 128, NT / 128), 256, 0, stream>>>(h2b, s2_b, t2, FSs, Cc);
    silu_mul_kernel<<<(NT * FSs + 255) / 256, 256, 0, stream>>>(t1, t2, heb, NT * FSs);
    gemm_bf16_kernel<1><<<dim3(Cc / 128, NT / 128), 256, 0, stream>>>(heb, sp_b, yac, Cc, FSs);

    // out = x1 + (y + z)
    final_add_kernel<<<(NT * Cc + 255) / 256, 256, 0, stream>>>(x1, yac, out, NT * Cc);
}
